// SECONDLikeModel_62380105008181
// MI455X (gfx1250) — matrix operand order repaired and checked by reading, not compiled, not run on hardware
//
#include <hip/hip_runtime.h>

// ---------------------------------------------------------------------------
// Types for CDNA5 WMMA (wave32): bf16 A/B fragments (16 x bf16 = 8 VGPRs),
// f32 C/D accumulator (8 VGPRs).
// ---------------------------------------------------------------------------
typedef __attribute__((ext_vector_type(16))) __bf16 v16bf;
typedef __attribute__((ext_vector_type(8)))  float  v8f;
typedef __attribute__((ext_vector_type(4)))  unsigned u32x4;

union FragU { unsigned u[8]; u32x4 q[2]; v16bf v; };

__device__ __forceinline__ unsigned short f2bf(float f) {
  unsigned u = __float_as_uint(f);
  u += 0x7fffu + ((u >> 16) & 1u);     // round-to-nearest-even
  return (unsigned short)(u >> 16);
}
__device__ __forceinline__ float bf2f(unsigned short h) {
  return __uint_as_float(((unsigned)h) << 16);
}
__device__ __forceinline__ unsigned packbf(float a, float b) {
  return (unsigned)f2bf(a) | ((unsigned)f2bf(b) << 16);
}

// Low 32 bits of a generic pointer into LDS == LDS byte offset (flat aperture
// truncation per CDNA5 ISA 10.2).
__device__ __forceinline__ unsigned lds_off(const void* p) {
  return (unsigned)(unsigned long long)p;
}

// gfx1250 async copy: per-lane 16B global -> LDS, tracked by ASYNCcnt.
__device__ __forceinline__ void async_ld_b128(unsigned ldsoff, const void* src) {
  asm volatile("global_load_async_to_lds_b128 %0, %1, off"
               :: "v"(ldsoff), "v"((unsigned long long)(__SIZE_TYPE__)src)
               : "memory");
}
__device__ __forceinline__ void wait_async0() {
#if __has_builtin(__builtin_amdgcn_s_wait_asynccnt)
  __builtin_amdgcn_s_wait_asynccnt(0);
#else
  asm volatile("s_wait_asynccnt 0x0" ::: "memory");
#endif
}

// ---------------------------------------------------------------------------
// Prep: mask0 = (active == 0); x0 = features * mask0, channel-last bf16
// packed as u32 pairs: x0[v*2+0] = {c0,c1}, x0[v*2+1] = {c2,c3}.
// ---------------------------------------------------------------------------
__global__ void prep_features_kernel(const float* __restrict__ feat,
                                     const int* __restrict__ active,
                                     unsigned* __restrict__ x0,
                                     unsigned char* __restrict__ mask0,
                                     int DHW, long total) {
  long v = (long)blockIdx.x * blockDim.x + threadIdx.x;
  if (v >= total) return;
  int b   = (int)(v / DHW);
  int rem = (int)(v % DHW);
  int act = (active[v] == 0);
  mask0[v] = (unsigned char)act;
  float m = act ? 1.f : 0.f;
  const float* fb = feat + ((long)b * 4) * DHW + rem;
  float f0 = fb[0] * m;
  float f1 = fb[(long)DHW] * m;
  float f2 = fb[2L * DHW] * m;
  float f3 = fb[3L * DHW] * m;
  x0[v * 2 + 0] = packbf(f0, f1);
  x0[v * 2 + 1] = packbf(f2, f3);
}

// ---------------------------------------------------------------------------
// Dilating mask downsample (stride 2, pad 1, 3x3x3 OR) + active-voxel count.
// ---------------------------------------------------------------------------
__global__ void make_mask_kernel(const unsigned char* __restrict__ msrc,
                                 unsigned char* __restrict__ mout,
                                 float* __restrict__ cnt,
                                 int Di, int Hi, int Wi,
                                 int Do, int Ho, int Wo, int total) {
  int v = blockIdx.x * blockDim.x + threadIdx.x;
  if (v >= total) return;
  int w = v % Wo; int t = v / Wo;
  int h = t % Ho; t /= Ho;
  int d = t % Do; int b = t / Do;
  int any = 0;
  for (int kd = 0; kd < 3; ++kd) {
    int di = 2 * d + kd - 1; if (di < 0 || di >= Di) continue;
    for (int kh = 0; kh < 3; ++kh) {
      int hi = 2 * h + kh - 1; if (hi < 0 || hi >= Hi) continue;
      const unsigned char* row = msrc + (((long)b * Di + di) * Hi + hi) * (long)Wi;
      for (int kw = 0; kw < 3; ++kw) {
        int wi = 2 * w + kw - 1; if (wi < 0 || wi >= Wi) continue;
        any |= row[wi];
      }
    }
  }
  mout[v] = (unsigned char)(any ? 1 : 0);
  unsigned long long bal = __ballot(any != 0);        // wave32: low 32 bits
  if ((threadIdx.x & 31) == 0)
    atomicAdd(cnt, (float)__popcll(bal));
}

// ---------------------------------------------------------------------------
// Weight prepack: [O][I][3][3][3] f32 -> chunk-major transposed bf16x2:
//   wp[((ck*COUT)+co)*20 + pl],  K = tap*CIN + cin, pair p = K/2,
//   ck = p>>4, pl = p&15, rows padded to 20 u32 (bank-friendly, 16B aligned).
// Identical layout in LDS -> weight chunks stage with async b128 copies.
// ---------------------------------------------------------------------------
template <int CIN, int COUT>
__global__ void prep_weights_kernel(const float* __restrict__ w,
                                    unsigned* __restrict__ wp) {
  constexpr int KP = ((27 * CIN + 31) / 32) * 32;
  int idx = blockIdx.x * blockDim.x + threadIdx.x;
  if (idx >= (KP / 2) * COUT) return;
  int p  = idx / COUT;
  int co = idx % COUT;
  int k0 = 2 * p, k1 = 2 * p + 1;
  float v0 = 0.f, v1 = 0.f;
  if (k0 < 27 * CIN) v0 = w[((long)co * CIN + (k0 % CIN)) * 27 + (k0 / CIN)];
  if (k1 < 27 * CIN) v1 = w[((long)co * CIN + (k1 % CIN)) * 27 + (k1 / CIN)];
  int ck = p >> 4, pl = p & 15;
  wp[((long)ck * COUT + co) * 20 + pl] = packbf(v0, v1);
}

__global__ void zero_f32_kernel(float* p, int n) {
  int i = blockIdx.x * blockDim.x + threadIdx.x;
  if (i < n) p[i] = 0.f;
}

// ---------------------------------------------------------------------------
// Implicit-GEMM conv3d 3x3x3 via v_wmma_f32_16x16x32_bf16.
// Block = (COUT/16) waves; wave cg owns channels [16cg,16cg+16).
// N-tiled: each wave computes NT WMMAs (NT*16 voxels along W) per weight
// fragment -> weight LDS traffic amortized NT-fold.
// Double-buffered LDS, all global->LDS staging overlapped with WMMA:
//   - weights: contiguous chunk via global_load_async_to_lds_b128
//   - activations (CIN>=32): k-chunk == one (tap, 32-ch block) => 64B/voxel
//     contiguous rows, staged with per-lane async b128 (halo zero-filled)
//   - activations (CIN==4): scalar gather, pair index hoisted per thread
// Row-transposed LDS tiles (pad 20 u32) => 2 x ds_load_b128 per fragment.
// BN sum/sumsq fused into the epilogue (register-batched over N-subtiles).
// ---------------------------------------------------------------------------
template <int CIN, int COUT, int STRIDE, int NT>
__global__ __launch_bounds__((COUT / 16) * 32)
void conv_wmma_kernel(const unsigned short* __restrict__ x,   // ch-last bf16
                      const unsigned* __restrict__ wp,        // [ck][COUT][20]
                      const unsigned char* __restrict__ maskOut,
                      unsigned short* __restrict__ y,         // ch-last bf16
                      float* __restrict__ gsum, float* __restrict__ gsq,
                      int Di, int Hi, int Wi, int Do, int Ho, int Wo) {
  constexpr int KP      = ((27 * CIN + 31) / 32) * 32;
  constexpr int NCHUNK  = KP / 32;
  constexpr int NWAVE   = COUT / 16;
  constexpr int THREADS = NWAVE * 32;

  __shared__ __align__(16) unsigned aBuf[2][NT * 16 * 20]; // [voxel][pair(+pad)]
  __shared__ __align__(16) unsigned wBuf[2][COUT * 20];    // [cout][pair(+pad)]
  __shared__ float lsum[COUT], lsq[COUT];

  const int tid  = threadIdx.x;
  const int lane = tid & 31;
  const int cg   = tid >> 5;

  const int w0 = blockIdx.x * (16 * NT);       // first output voxel along W
  const int ho = blockIdx.y;
  const int b  = blockIdx.z / Do;
  const int dz = blockIdx.z % Do;

  for (int i = tid; i < COUT; i += THREADS) { lsum[i] = 0.f; lsq[i] = 0.f; }

  const unsigned* xw = (const unsigned*)x;     // bf16-pair view (CIN even)

  auto stage = [&](int ck, int buf) {
    unsigned abase = lds_off(&aBuf[buf][0]);
    if constexpr (CIN >= 32) {
      // Whole chunk is one (tap, 32-channel block): 64B contiguous per voxel.
      int t  = (ck * 32) / CIN;
      int c  = (ck * 32) % CIN;
      int kd = t / 9, kh = (t / 3) % 3, kw = t % 3;
      int di = STRIDE * dz + kd - 1;
      int hi = STRIDE * ho + kh - 1;
      bool dhok = (di >= 0) && (di < Di) && (hi >= 0) && (hi < Hi);
      const char* rowbase = (const char*)x +
          ((((long)b * Di + di) * Hi + hi) * (long)Wi * CIN + c) * 2;
      for (int e = tid; e < NT * 16 * 4; e += THREADS) {
        int n = e >> 2, j = e & 3;              // voxel, 16B segment
        int wi = STRIDE * (w0 + n) + kw - 1;
        if (dhok && wi >= 0 && wi < Wi) {
          async_ld_b128(abase + (unsigned)(n * 20 + 4 * j) * 4,
                        rowbase + (long)wi * CIN * 2 + j * 16);
        } else {
          u32x4 z = {0u, 0u, 0u, 0u};
          *(u32x4*)&aBuf[buf][n * 20 + 4 * j] = z;
        }
      }
    } else {
      // CIN==4: pairs straddle taps; per-thread pair index is loop-invariant.
      int p = tid & 15;
      int k = ck * 32 + 2 * p;
      int t = k / CIN;
      int c = k % CIN;
      int kd = t / 9, kh = (t / 3) % 3, kw = t % 3;
      int di = STRIDE * dz + kd - 1;
      int hi = STRIDE * ho + kh - 1;
      bool dhok = (t < 27) && di >= 0 && di < Di && hi >= 0 && hi < Hi;
      const unsigned* src = xw +
          (((long)b * Di + di) * Hi + hi) * (long)Wi * (CIN / 2) + (c >> 1);
      for (int n = tid >> 4; n < NT * 16; n += THREADS / 16) {
        int wi = STRIDE * (w0 + n) + kw - 1;
        unsigned val = 0u;
        if (dhok && wi >= 0 && wi < Wi) val = src[(long)wi * (CIN / 2)];
        aBuf[buf][n * 20 + p] = val;
      }
    }
    // Weight chunk: straight contiguous copy, async global -> LDS (16B/lane).
    const char* src = (const char*)(wp + (long)ck * COUT * 20);
    unsigned dst = lds_off(&wBuf[buf][0]);
    for (int i = tid; i < COUT * 5; i += THREADS)   // COUT*20 u32 / 4 per b128
      async_ld_b128(dst + i * 16, src + i * 16);
  };

  v8f acc[NT];
#pragma unroll
  for (int s = 0; s < NT; ++s) acc[s] = (v8f){0.f,0.f,0.f,0.f,0.f,0.f,0.f,0.f};

  // Fragment lane mapping (wave32 ISA layouts).
  const int n     = lane & 15;                 // B/D column (voxel-in-subtile)
  const int co0   = cg * 16 + (lane & 15);     // A row (cout)
  const int base4 = (lane >> 4) * 4;           // A k-pair half select
  const int baseP = (lane >> 4) * 8;           // unused; B also uses base4

  stage(0, 0);
  wait_async0();
  for (int ck = 0; ck < NCHUNK; ++ck) {
    __syncthreads();
    if (ck + 1 < NCHUNK) stage(ck + 1, (ck + 1) & 1);
    const unsigned* wb = wBuf[ck & 1] + co0 * 20;
    FragU fa;
    fa.q[0] = *(const u32x4*)(wb + base4);         // pairs base4..base4+3
    fa.q[1] = *(const u32x4*)(wb + 8 + base4);     // pairs 8+base4..11+base4
#pragma unroll
    for (int s = 0; s < NT; ++s) {                 // reuse A across N-subtiles
      const unsigned* ab = aBuf[ck & 1] + (s * 16 + n) * 20;
      FragU fb;
      fb.q[0] = *(const u32x4*)(ab + base4);       // pairs base4..base4+3
      fb.q[1] = *(const u32x4*)(ab + 8 + base4);   // pairs 8+base4..11+base4
      acc[s] = __builtin_amdgcn_wmma_f32_16x16x32_bf16(false, fa.v, false, fb.v,
                                                       (short)0, acc[s],
                                                       false, false);
    }
    wait_async0();   // drain next chunk's async copies after this WMMA group
  }

  // Epilogue: packed 16B stores of raw conv output + masked BN statistics.
  const int  mhi     = (lane >> 4) * 8;
  const long rowBase = (((long)b * Do + dz) * Ho + ho) * (long)Wo;
  float psum[8], psq[8];
#pragma unroll
  for (int r = 0; r < 8; ++r) { psum[r] = 0.f; psq[r] = 0.f; }
#pragma unroll
  for (int s = 0; s < NT; ++s) {
    const int  wo     = w0 + s * 16 + n;
    const long voxIdx = rowBase + wo;
    const float msk   = maskOut[voxIdx] ? 1.f : 0.f;
    unsigned* yp = (unsigned*)(y + voxIdx * COUT) + (cg * 16 + mhi) / 2;
    u32x4 pk;
    pk.x = packbf(acc[s][0], acc[s][1]);
    pk.y = packbf(acc[s][2], acc[s][3]);
    pk.z = packbf(acc[s][4], acc[s][5]);
    pk.w = packbf(acc[s][6], acc[s][7]);
    *(u32x4*)yp = pk;
    if (msk != 0.f) {
#pragma unroll
      for (int r = 0; r < 8; ++r) {
        float v = acc[s][r];
        psum[r] += v;
        psq[r]  += v * v;
      }
    }
  }
#pragma unroll
  for (int r = 0; r < 8; ++r) {
    int co = cg * 16 + mhi + r;
    atomicAdd(&lsum[co], psum[r]);
    atomicAdd(&lsq[co],  psq[r]);
  }
  __syncthreads();
  for (int i = tid; i < COUT; i += THREADS) {
    atomicAdd(&gsum[i], lsum[i]);
    atomicAdd(&gsq[i],  lsq[i]);
  }
}

// ---------------------------------------------------------------------------
// Masked BN + ReLU, in place, bf16-pair (u32) granularity.
// ---------------------------------------------------------------------------
template <int COUT>
__global__ void bn_relu_kernel(unsigned short* __restrict__ y,
                               const unsigned char* __restrict__ mask,
                               const float* __restrict__ gsum,
                               const float* __restrict__ gsq,
                               const float* __restrict__ cnt,
                               const float* __restrict__ gamma,
                               const float* __restrict__ beta,
                               long nvox) {
  long v = (long)blockIdx.x * blockDim.x + threadIdx.x;
  if (v >= nvox) return;
  const float inv = 1.f / cnt[0];
  const float msk = mask[v] ? 1.f : 0.f;
  unsigned* yp = (unsigned*)(y + v * COUT);
#pragma unroll 4
  for (int c2 = 0; c2 < COUT / 2; ++c2) {
    unsigned u = yp[c2];
    int ca = 2 * c2, cb = ca + 1;
    float ma = gsum[ca] * inv, mb = gsum[cb] * inv;
    float ra = rsqrtf(gsq[ca] * inv - ma * ma + 1e-5f);
    float rb = rsqrtf(gsq[cb] * inv - mb * mb + 1e-5f);
    float va = (bf2f((unsigned short)u)         - ma) * ra * gamma[ca] + beta[ca];
    float vb = (bf2f((unsigned short)(u >> 16)) - mb) * rb * gamma[cb] + beta[cb];
    va = fmaxf(va, 0.f) * msk;
    vb = fmaxf(vb, 0.f) * msk;
    yp[c2] = packbf(va, vb);
  }
}

// ---------------------------------------------------------------------------
// Per-voxel heads: 128 -> 3 logits + 7 bbox regs, masked; NCDHW output.
// ---------------------------------------------------------------------------
__global__ void heads_kernel(const unsigned short* __restrict__ x3,
                             const unsigned char* __restrict__ mask3,
                             const float* __restrict__ Wc, const float* __restrict__ bc,
                             const float* __restrict__ Wr, const float* __restrict__ br,
                             float* __restrict__ out) {
  __shared__ float WL[10 * 128];
  __shared__ float bL[10];
  int tid = threadIdx.x;
  for (int i = tid; i < 3 * 128; i += 256) WL[i] = Wc[i];
  for (int i = tid; i < 7 * 128; i += 256) WL[3 * 128 + i] = Wr[i];
  if (tid < 3)       bL[tid] = bc[tid];
  else if (tid < 10) bL[tid] = br[tid - 3];
  __syncthreads();

  int v = blockIdx.x * 256 + tid;              // 0 .. 81919 (exact grid)
  const int SPAT = 10 * 64 * 64;               // 40960
  int b = v / SPAT;
  int s = v % SPAT;
  float acc[10];
#pragma unroll
  for (int k = 0; k < 10; ++k) acc[k] = 0.f;
  if (mask3[v]) {
    const unsigned* xp = (const unsigned*)(x3 + (long)v * 128);
    for (int c2 = 0; c2 < 64; ++c2) {
      unsigned u = xp[c2];
      float xa = bf2f((unsigned short)u);
      float xb = bf2f((unsigned short)(u >> 16));
#pragma unroll
      for (int k = 0; k < 10; ++k)
        acc[k] += WL[k * 128 + 2 * c2] * xa + WL[k * 128 + 2 * c2 + 1] * xb;
    }
#pragma unroll
    for (int k = 0; k < 10; ++k) acc[k] += bL[k];
  }
  float* logits = out;
  float* bboxes = out + 2L * 3 * SPAT;
#pragma unroll
  for (int k = 0; k < 3; ++k) logits[((long)b * 3 + k) * SPAT + s] = acc[k];
#pragma unroll
  for (int k = 0; k < 7; ++k) bboxes[((long)b * 7 + k) * SPAT + s] = acc[3 + k];
}

// ---------------------------------------------------------------------------
// Launcher
// ---------------------------------------------------------------------------
extern "C" void kernel_launch(void* const* d_in, const int* in_sizes, int n_in,
                              void* d_out, int out_size, void* d_ws, size_t ws_size,
                              hipStream_t stream) {
  (void)in_sizes; (void)n_in; (void)out_size; (void)ws_size;
  const float* features = (const float*)d_in[0];
  const int*   active   = (const int*)d_in[1];
  const float* w1 = (const float*)d_in[2];
  const float* g1 = (const float*)d_in[3];
  const float* b1 = (const float*)d_in[4];
  const float* w2 = (const float*)d_in[5];
  const float* g2 = (const float*)d_in[6];
  const float* b2 = (const float*)d_in[7];
  const float* w3 = (const float*)d_in[8];
  const float* g3 = (const float*)d_in[9];
  const float* b3 = (const float*)d_in[10];
  const float* Wc = (const float*)d_in[11];
  const float* bc = (const float*)d_in[12];
  const float* Wr = (const float*)d_in[13];
  const float* br = (const float*)d_in[14];
  float* out = (float*)d_out;

  const long NV0 = 2L * 40 * 256 * 256;   // 5,242,880
  const long NV1 = 2L * 20 * 128 * 128;   //   655,360
  const long NV3 = 2L * 10 * 64 * 64;     //    81,920

  char* base = (char*)d_ws;
  size_t off = 0;
  auto alloc = [&](size_t bytes) -> char* {
    char* p = base + off;
    off += (bytes + 255) & ~(size_t)255;
    return p;
  };

  unsigned*       x0    = (unsigned*)alloc(NV0 * 2 * sizeof(unsigned));
  unsigned char*  mask0 = (unsigned char*)alloc(NV0);
  unsigned char*  mask1 = (unsigned char*)alloc(NV1);
  unsigned char*  mask3 = (unsigned char*)alloc(NV3);
  unsigned short* y1    = (unsigned short*)alloc(NV1 * 32 * 2);
  unsigned short* y2    = (unsigned short*)alloc(NV1 * 64 * 2);
  unsigned short* y3    = (unsigned short*)alloc(NV3 * 128 * 2);
  unsigned* wp1 = (unsigned*)alloc(4L  * 32  * 20 * 4);   // [ck][COUT][20]
  unsigned* wp2 = (unsigned*)alloc(27L * 64  * 20 * 4);
  unsigned* wp3 = (unsigned*)alloc(54L * 128 * 20 * 4);
  float* stats = (float*)alloc(450 * 4);
  float* s1 = stats;        float* q1 = stats + 32;
  float* s2 = stats + 64;   float* q2 = stats + 128;
  float* s3 = stats + 192;  float* q3 = stats + 320;
  float* cnt1 = stats + 448; float* cnt3 = stats + 449;

  zero_f32_kernel<<<1, 512, 0, stream>>>(stats, 450);
  prep_features_kernel<<<(int)(NV0 / 256), 256, 0, stream>>>(
      features, active, x0, mask0, 40 * 256 * 256, NV0);
  make_mask_kernel<<<(int)(NV1 / 256), 256, 0, stream>>>(
      mask0, mask1, cnt1, 40, 256, 256, 20, 128, 128, (int)NV1);
  make_mask_kernel<<<(int)(NV3 / 256), 256, 0, stream>>>(
      mask1, mask3, cnt3, 20, 128, 128, 10, 64, 64, (int)NV3);
  prep_weights_kernel<4, 32><<<(64 * 32 + 255) / 256, 256, 0, stream>>>(w1, wp1);
  prep_weights_kernel<32, 64><<<(432 * 64 + 255) / 256, 256, 0, stream>>>(w2, wp2);
  prep_weights_kernel<64, 128><<<(864 * 128 + 255) / 256, 256, 0, stream>>>(w3, wp3);

  // Block 1: regular sparse conv, stride 2, 4 -> 32  (64-voxel N tile)
  conv_wmma_kernel<4, 32, 2, 4><<<dim3(2, 128, 2 * 20), 64, 0, stream>>>(
      (const unsigned short*)x0, wp1, mask1, y1, s1, q1,
      40, 256, 256, 20, 128, 128);
  bn_relu_kernel<32><<<(int)(NV1 / 256), 256, 0, stream>>>(
      y1, mask1, s1, q1, cnt1, g1, b1, NV1);

  // Block 2: submanifold conv, stride 1, 32 -> 64
  conv_wmma_kernel<32, 64, 1, 4><<<dim3(2, 128, 2 * 20), 128, 0, stream>>>(
      y1, wp2, mask1, y2, s2, q2, 20, 128, 128, 20, 128, 128);
  bn_relu_kernel<64><<<(int)(NV1 / 256), 256, 0, stream>>>(
      y2, mask1, s2, q2, cnt1, g2, b2, NV1);

  // Block 3: regular sparse conv, stride 2, 64 -> 128
  conv_wmma_kernel<64, 128, 2, 4><<<dim3(1, 64, 2 * 10), 256, 0, stream>>>(
      y2, wp3, mask3, y3, s3, q3, 20, 128, 128, 10, 64, 64);
  bn_relu_kernel<128><<<(int)(NV3 / 256), 256, 0, stream>>>(
      y3, mask3, s3, q3, cnt3, g3, b3, NV3);

  // Heads
  heads_kernel<<<(int)(NV3 / 256), 256, 0, stream>>>(
      y3, mask3, Wc, bc, Wr, br, out);
}
